// VolumeAttention_68298569941600
// MI455X (gfx1250) — compile-verified, matrix operand order checked by reading, not run on hardware
//
#include <hip/hip_runtime.h>
#include <hip/hip_bf16.h>

// ---------------- constants ----------------
#define C_    256
#define H_    8
#define HD_   32
#define D_    6
#define NQ_   100
#define B_    4
#define NKMAX 4096

#define USE_ASYNC_COPY 1

typedef __attribute__((ext_vector_type(16))) __bf16 v16bf;
typedef __attribute__((ext_vector_type(8)))  __bf16 v8bf;
typedef __attribute__((ext_vector_type(4)))  __bf16 v4bf;
typedef __attribute__((ext_vector_type(8)))  float  v8f;

// Fragment layout for 16-bit WMMA operands (ISA 7.12.2): lane's 16 elements are
// two contiguous 8-element runs: k = half*8 .. +7 and k = 16+half*8 .. +7.
__device__ __forceinline__ v16bf frag_ld(const __bf16* row, int half) {
    v8bf lo = *(const v8bf*)(row + (half << 3));
    v8bf hi = *(const v8bf*)(row + 16 + (half << 3));
    return __builtin_shufflevector(lo, hi, 0, 1, 2, 3, 4, 5, 6, 7,
                                   8, 9, 10, 11, 12, 13, 14, 15);
}

__device__ __forceinline__ int wk_idx(int e, int half) {
    int j = e >> 1, lo = e & 1;
    int base = half << 3;
    return (j < 4 ? base + 2 * j : 16 + base + 2 * (j - 4)) + lo;
}

#if USE_ASYNC_COPY
__device__ __forceinline__ void async_b128(const void* gp, void* lp) {
    unsigned loff = (unsigned)(size_t)lp;
    unsigned long long ga = (unsigned long long)(size_t)gp;
    asm volatile("global_load_async_to_lds_b128 %0, %1, off"
                 :: "v"(loff), "v"(ga) : "memory");
}
__device__ __forceinline__ void async_b64(const void* gp, void* lp) {
    unsigned loff = (unsigned)(size_t)lp;
    unsigned long long ga = (unsigned long long)(size_t)gp;
    asm volatile("global_load_async_to_lds_b64 %0, %1, off"
                 :: "v"(loff), "v"(ga) : "memory");
}
__device__ __forceinline__ void async_wait0() {
    asm volatile("s_wait_asynccnt 0x0" ::: "memory");
}
#endif

// ---------------- weight prep ----------------
__global__ __launch_bounds__(256) void conv_scale_bf16(const float* __restrict__ w,
                                                       const float* __restrict__ g,
                                                       __bf16* __restrict__ out,
                                                       long long n, int N) {
    long long i = (long long)blockIdx.x * blockDim.x + threadIdx.x;
    if (i >= n) return;
    float v = w[i];
    if (g) {
        int k = (int)((i / N) % C_);
        int d = (int)(i / ((long long)N * C_));
        v *= g[d * C_ + k];
    }
    out[i] = (__bf16)v;
}

__global__ __launch_bounds__(256) void fold_bias(const float* __restrict__ bvec,
                                                 const float* __restrict__ w,
                                                 float* __restrict__ out) {
    int idx = blockIdx.x * blockDim.x + threadIdx.x;
    if (idx >= D_ * C_) return;
    int d = idx / C_, n = idx % C_;
    float s = 0.f;
    const float* bp = bvec + (size_t)d * C_;
    const float* wp = w + (size_t)d * C_ * C_ + n;
    for (int k = 0; k < C_; ++k) s += bp[k] * wp[(size_t)k * C_];
    out[idx] = s;
}

// ---------------- tokenize + normalize ----------------
__global__ __launch_bounds__(256) void feat_norm(const float* __restrict__ f,
                                                 __bf16* __restrict__ out, int hw2) {
    int token = blockIdx.x;
    int b = token / hw2, t = token % hw2;
    int c = threadIdx.x;
    float x = f[((size_t)b * C_ + c) * hw2 + t];
    __shared__ float red[256];
    red[c] = x; __syncthreads();
    for (int s = 128; s > 0; s >>= 1) { if (c < s) red[c] += red[c + s]; __syncthreads(); }
    float m = red[0] * (1.f / C_); __syncthreads();
    float dv = x - m;
    red[c] = dv * dv; __syncthreads();
    for (int s = 128; s > 0; s >>= 1) { if (c < s) red[c] += red[c + s]; __syncthreads(); }
    float var = red[0] * (1.f / C_);
    out[(size_t)token * C_ + c] = (__bf16)(dv * rsqrtf(var + 1e-5f));
}

__global__ __launch_bounds__(256) void ln_rows(const float* __restrict__ in,
                                               __bf16* __restrict__ out) {
    int row = blockIdx.x;
    int c = threadIdx.x;
    float x = in[(size_t)row * C_ + c];
    __shared__ float red[256];
    red[c] = x; __syncthreads();
    for (int s = 128; s > 0; s >>= 1) { if (c < s) red[c] += red[c + s]; __syncthreads(); }
    float m = red[0] * (1.f / C_); __syncthreads();
    float dv = x - m;
    red[c] = dv * dv; __syncthreads();
    for (int s = 128; s > 0; s >>= 1) { if (c < s) red[c] += red[c + s]; __syncthreads(); }
    float var = red[0] * (1.f / C_);
    out[(size_t)row * C_ + c] = (__bf16)(dv * rsqrtf(var + 1e-5f));
}

__global__ __launch_bounds__(256) void cast_bf16(const float* __restrict__ src,
                                                 __bf16* __restrict__ dst, int n) {
    int i = blockIdx.x * blockDim.x + threadIdx.x;
    if (i < n) dst[i] = (__bf16)src[i];
}

__global__ __launch_bounds__(256) void init_q(const float* __restrict__ qe,
                                              float* __restrict__ q, int n) {
    int i = blockIdx.x * blockDim.x + threadIdx.x;
    if (i < n) q[i] = qe[i % (NQ_ * C_)];
}

// ---------------- WMMA GEMM: Y[M,N] = A[M,256]@W[256,N] + bias ----------------
// Block 256 thr = 8 waves (2 Mwaves x 4 Nwaves); block tile 64x128,
// wave tile 32x32 = 4 accumulators of v_wmma_f32_16x16x32_bf16.
// A tile staged via async-LDS copy; W tile staged transposed for contiguous frags.
__global__ __launch_bounds__(256) void gemm_bf16(const __bf16* __restrict__ A,
                                                 const __bf16* __restrict__ W,
                                                 const float* __restrict__ bias,
                                                 float* __restrict__ Yf,
                                                 __bf16* __restrict__ Yb,
                                                 int M, int N, int K) {
    __shared__ __bf16 As[64][32];
    __shared__ __bf16 Wt[128][32];   // [col][k]
    int tid = threadIdx.x;
    int lane = tid & 31, wid = tid >> 5;
    int wm = wid >> 2, wn = wid & 3;
    int bm = blockIdx.x * 64, bn = blockIdx.y * 128;
    int half = lane >> 4, l16 = lane & 15;
    v8f acc00 = {}, acc01 = {}, acc10 = {}, acc11 = {};
    for (int k0 = 0; k0 < K; k0 += 32) {
        { // stage A tile 64x32 (contiguous rows)
            int r = tid >> 2, c = (tid & 3) * 8;
            int gr = bm + r;
            const __bf16* gp = A + (size_t)gr * K + k0 + c;
            __bf16* lp = &As[r][c];
#if USE_ASYNC_COPY
            if (gr < M) async_b128(gp, lp);
            else        *(uint4*)lp = make_uint4(0u, 0u, 0u, 0u);
#else
            if (gr < M) { *(uint4*)lp = *(const uint4*)gp;
                          if (k0 + 32 < K) __builtin_prefetch(gp + 32, 0, 1); }
            else        *(uint4*)lp = make_uint4(0u, 0u, 0u, 0u);
#endif
        }
        { // stage W tile 32x128, transposed into Wt[col][k]
            int r = tid & 31;            // k row
            int c = (tid >> 5) * 16;     // col group of 16
            const __bf16* gp = W + (size_t)(k0 + r) * N + bn + c;
            v8bf w0 = *(const v8bf*)gp;
            v8bf w1 = *(const v8bf*)(gp + 8);
#pragma unroll
            for (int j = 0; j < 8; ++j) {
                Wt[c + j][r]     = w0[j];
                Wt[c + 8 + j][r] = w1[j];
            }
        }
#if USE_ASYNC_COPY
        async_wait0();
#endif
        __syncthreads();
        int arow = wm * 32 + l16;
        int bcol = wn * 32 + l16;
        v16bf af0 = frag_ld(&As[arow][0], half);
        v16bf af1 = frag_ld(&As[arow + 16][0], half);
        v16bf bf0 = frag_ld(&Wt[bcol][0], half);
        v16bf bf1 = frag_ld(&Wt[bcol + 16][0], half);
        acc00 = __builtin_amdgcn_wmma_f32_16x16x32_bf16(false, af0, false, bf0,
                                                        (short)0, acc00, false, false);
        acc01 = __builtin_amdgcn_wmma_f32_16x16x32_bf16(false, af0, false, bf1,
                                                        (short)0, acc01, false, false);
        acc10 = __builtin_amdgcn_wmma_f32_16x16x32_bf16(false, af1, false, bf0,
                                                        (short)0, acc10, false, false);
        acc11 = __builtin_amdgcn_wmma_f32_16x16x32_bf16(false, af1, false, bf1,
                                                        (short)0, acc11, false, false);
        __syncthreads();
    }
    int gn0 = bn + wn * 32 + l16;
    int gn1 = gn0 + 16;
    float bv0 = bias ? bias[gn0] : 0.f;
    float bv1 = bias ? bias[gn1] : 0.f;
#pragma unroll
    for (int r = 0; r < 8; ++r) {
        int gm0 = bm + wm * 32 + r + (half << 3);
        int gm1 = gm0 + 16;
        if (gm0 < M) {
            float a = acc00[r] + bv0, b = acc01[r] + bv1;
            if (Yf) { Yf[(size_t)gm0 * N + gn0] = a; Yf[(size_t)gm0 * N + gn1] = b; }
            if (Yb) { Yb[(size_t)gm0 * N + gn0] = (__bf16)a; Yb[(size_t)gm0 * N + gn1] = (__bf16)b; }
        }
        if (gm1 < M) {
            float a = acc10[r] + bv0, b = acc11[r] + bv1;
            if (Yf) { Yf[(size_t)gm1 * N + gn0] = a; Yf[(size_t)gm1 * N + gn1] = b; }
            if (Yb) { Yb[(size_t)gm1 * N + gn0] = (__bf16)a; Yb[(size_t)gm1 * N + gn1] = (__bf16)b; }
        }
    }
}

// ---------------- flash-style multi-head attention (WMMA) ----------------
__global__ __launch_bounds__(256) void mha_flash(const __bf16* __restrict__ Qb, int ldq, long long qbs,
                                                 const __bf16* __restrict__ Kb, int ldk, long long kbs,
                                                 const __bf16* __restrict__ Vb, int ldv, long long vbs,
                                                 __bf16* __restrict__ Ob, int ldo, long long obs,
                                                 int nq, int nk, float scale) {
    __shared__ __bf16 Ks[32][32];     // [key][dim]
    __shared__ __bf16 Vt[32][32];     // [dim][key] (transposed)
    __shared__ __bf16 Ps[8][16][32];  // per-wave P tile
    int tid = threadIdx.x, lane = tid & 31, wid = tid >> 5;
    int b = blockIdx.x / H_, h = blockIdx.x % H_;
    int ntiles = (nq + 15) >> 4;
    int qt = wid < ntiles ? wid : ntiles - 1;
    int half = lane >> 4, l16 = lane & 15;

    v16bf aq;
    int qrow = qt * 16 + l16;
#pragma unroll
    for (int e = 0; e < 16; ++e) {
        int kk = wk_idx(e, half);
        aq[e] = (qrow < nq) ? Qb[qbs * b + (long long)qrow * ldq + h * HD_ + kk] : (__bf16)0.f;
    }

    v8f o0 = {}, o1 = {};
    float m_i[8], l_i[8];
#pragma unroll
    for (int r = 0; r < 8; ++r) { m_i[r] = -1e30f; l_i[r] = 0.f; }

    for (int kc = 0; kc < nk; kc += 32) {
        { // stage: K rows async (contiguous), V transposed scatter
            int key = tid >> 3, c4 = (tid & 7) * 4;
            int gk = kc + key;
            if (gk < nk) {
                const __bf16* kp = Kb + kbs * b + (long long)gk * ldk + h * HD_ + c4;
                const __bf16* vp = Vb + vbs * b + (long long)gk * ldv + h * HD_ + c4;
#if USE_ASYNC_COPY
                async_b64(kp, &Ks[key][c4]);
#else
                *(uint2*)&Ks[key][c4] = *(const uint2*)kp;
#endif
                v4bf vv = *(const v4bf*)vp;
#pragma unroll
                for (int j = 0; j < 4; ++j) Vt[c4 + j][key] = vv[j];
            } else {
                *(uint2*)&Ks[key][c4] = make_uint2(0u, 0u);
#pragma unroll
                for (int j = 0; j < 4; ++j) Vt[c4 + j][key] = (__bf16)0.f;
            }
        }
#if USE_ASYNC_COPY
        async_wait0();
#endif
        __syncthreads();

        v16bf bk0 = frag_ld(&Ks[l16][0], half);       // B[k=dim][n=key 0..15]
        v16bf bk1 = frag_ld(&Ks[l16 + 16][0], half);  // keys 16..31
        v8f zero = {};
        v8f s0 = __builtin_amdgcn_wmma_f32_16x16x32_bf16(false, aq, false, bk0,
                                                         (short)0, zero, false, false);
        v8f s1 = __builtin_amdgcn_wmma_f32_16x16x32_bf16(false, aq, false, bk1,
                                                         (short)0, zero, false, false);
        int kcol0 = kc + l16, kcol1 = kc + 16 + l16;
#pragma unroll
        for (int r = 0; r < 8; ++r) {
            float v0 = (kcol0 < nk) ? s0[r] * scale : -1e30f;
            float v1 = (kcol1 < nk) ? s1[r] * scale : -1e30f;
            float mx = fmaxf(v0, v1);
            for (int off = 1; off < 16; off <<= 1) mx = fmaxf(mx, __shfl_xor(mx, off, 16));
            float mnew = fmaxf(m_i[r], mx);
            float p0 = __expf(v0 - mnew);
            float p1 = __expf(v1 - mnew);
            float ps = p0 + p1;
            for (int off = 1; off < 16; off <<= 1) ps += __shfl_xor(ps, off, 16);
            float corr = __expf(m_i[r] - mnew);
            l_i[r] = l_i[r] * corr + ps;
            m_i[r] = mnew;
            o0[r] *= corr; o1[r] *= corr;
            int lr = r + (half << 3);
            Ps[wid][lr][l16]      = (__bf16)p0;
            Ps[wid][lr][l16 + 16] = (__bf16)p1;
        }
        __syncthreads();

        v16bf pa  = frag_ld(&Ps[wid][l16][0], half);  // A = P (16 q x 32 keys)
        v16bf bv0 = frag_ld(&Vt[l16][0], half);       // B = V (32 keys x dims 0..15)
        v16bf bv1 = frag_ld(&Vt[l16 + 16][0], half);  // dims 16..31
        o0 = __builtin_amdgcn_wmma_f32_16x16x32_bf16(false, pa, false, bv0,
                                                     (short)0, o0, false, false);
        o1 = __builtin_amdgcn_wmma_f32_16x16x32_bf16(false, pa, false, bv1,
                                                     (short)0, o1, false, false);
        __syncthreads();
    }

#pragma unroll
    for (int r = 0; r < 8; ++r) {
        int row = qt * 16 + r + (half << 3);
        if (row < nq && wid < ntiles) {
            float inv = 1.f / l_i[r];
            Ob[obs * b + (long long)row * ldo + h * HD_ + l16]      = (__bf16)(o0[r] * inv);
            Ob[obs * b + (long long)row * ldo + h * HD_ + 16 + l16] = (__bf16)(o1[r] * inv);
        }
    }
}

// ---------------- matching / softmax / mix / combine ----------------
__global__ __launch_bounds__(128) void match_scores(const float* __restrict__ q1,
                                                    const float* __restrict__ q2,
                                                    float* __restrict__ ms) {
    int b = blockIdx.x / NQ_, i = blockIdx.x % NQ_;
    __shared__ float qi[C_];
    for (int c = threadIdx.x; c < C_; c += blockDim.x)
        qi[c] = q1[((size_t)b * NQ_ + i) * C_ + c];
    __syncthreads();
    for (int j = threadIdx.x; j < NQ_; j += blockDim.x) {
        const float* r = q2 + ((size_t)b * NQ_ + j) * C_;
        float s = 0.f;
        for (int c = 0; c < C_; ++c) s += qi[c] * r[c];
        ms[((size_t)b * NQ_ + i) * NQ_ + j] = s;
    }
}

__global__ __launch_bounds__(128) void softmax_ms(const float* __restrict__ ms,
                                                  float* __restrict__ out, int axis) {
    int b = blockIdx.x / NQ_, r = blockIdx.x % NQ_;
    int t = threadIdx.x;
    size_t base; int stride;
    if (axis == 2) { base = ((size_t)b * NQ_ + r) * NQ_; stride = 1; }
    else           { base = (size_t)b * NQ_ * NQ_ + r;   stride = NQ_; }
    float v = (t < NQ_) ? ms[base + (size_t)t * stride] : -1e30f;
    __shared__ float red[128];
    red[t] = v; __syncthreads();
    for (int s = 64; s > 0; s >>= 1) { if (t < s) red[t] = fmaxf(red[t], red[t + s]); __syncthreads(); }
    float mx = red[0]; __syncthreads();
    float e = (t < NQ_) ? __expf(v - mx) : 0.f;
    red[t] = e; __syncthreads();
    for (int s = 64; s > 0; s >>= 1) { if (t < s) red[t] += red[t + s]; __syncthreads(); }
    float sum = red[0];
    if (t < NQ_) out[base + (size_t)t * stride] = e / sum;
}

__global__ __launch_bounds__(256) void attn_mix(const float* __restrict__ base,
                                                const float* __restrict__ src,
                                                const float* __restrict__ w,
                                                float* __restrict__ out, int mode) {
    int b = blockIdx.x / NQ_, r = blockIdx.x % NQ_;
    int c = threadIdx.x;
    __shared__ float wl[NQ_];
    size_t wo = (mode == 0) ? ((size_t)b * NQ_ + r) * NQ_ : ((size_t)b * NQ_ * NQ_ + r);
    int wstride = (mode == 0) ? 1 : NQ_;
    for (int j = c; j < NQ_; j += blockDim.x) wl[j] = w[wo + (size_t)j * wstride];
    __syncthreads();
    float acc = base[((size_t)b * NQ_ + r) * C_ + c];
    for (int j = 0; j < NQ_; ++j)
        acc += wl[j] * src[((size_t)b * NQ_ + j) * C_ + c];
    out[((size_t)b * NQ_ + r) * C_ + c] = acc;
}

__global__ __launch_bounds__(256) void combine_softmax(float* __restrict__ q,
                                                       const float* __restrict__ r1,
                                                       const float* __restrict__ r2) {
    int row = blockIdx.x;
    int c = threadIdx.x;
    size_t idx = (size_t)row * C_ + c;
    float v = q[idx] + r1[idx] + r2[idx];
    __shared__ float red[256];
    red[c] = v; __syncthreads();
    for (int s = 128; s > 0; s >>= 1) { if (c < s) red[c] = fmaxf(red[c], red[c + s]); __syncthreads(); }
    float mx = red[0]; __syncthreads();
    float e = __expf(v - mx);
    red[c] = e; __syncthreads();
    for (int s = 128; s > 0; s >>= 1) { if (c < s) red[c] += red[c + s]; __syncthreads(); }
    float sum = red[0];
    q[idx] = e / sum;
}

// ---------------- host orchestration ----------------
static inline size_t align256(size_t x) { return (x + 255) & ~(size_t)255; }

extern "C" void kernel_launch(void* const* d_in, const int* in_sizes, int n_in,
                              void* d_out, int out_size, void* d_ws, size_t ws_size,
                              hipStream_t stream) {
    (void)in_sizes; (void)n_in; (void)out_size; (void)ws_size;
    const float* feat1[4] = {(const float*)d_in[0], (const float*)d_in[1],
                             (const float*)d_in[2], (const float*)d_in[3]};
    const float* feat2[4] = {(const float*)d_in[4], (const float*)d_in[5],
                             (const float*)d_in[6], (const float*)d_in[7]};
    const float* query_embed = (const float*)d_in[8];
    const float* ca_ng = (const float*)d_in[9];
    const float* ca_nb = (const float*)d_in[10];
    const float* ca_wq = (const float*)d_in[11];
    const float* ca_wk = (const float*)d_in[12];
    const float* ca_wv = (const float*)d_in[13];
    const float* ca_wp = (const float*)d_in[14];
    const float* ca_bp = (const float*)d_in[15];
    const float* sa_wqkv = (const float*)d_in[16];
    const float* sa_wp = (const float*)d_in[17];
    const float* sa_bp = (const float*)d_in[18];

    char* base = (char*)d_ws;
    size_t off = 0;
    auto grab = [&](size_t bytes) { void* p = base + off; off = align256(off + bytes); return p; };
    const size_t WCC = (size_t)D_ * C_ * C_;
    __bf16* wqf   = (__bf16*)grab(WCC * 2);
    __bf16* wkf   = (__bf16*)grab(WCC * 2);
    __bf16* wvf   = (__bf16*)grab(WCC * 2);
    __bf16* wpf   = (__bf16*)grab(WCC * 2);
    __bf16* sawpf = (__bf16*)grab(WCC * 2);
    __bf16* sqkvf = (__bf16*)grab(WCC * 3 * 2);
    float*  bqf   = (float*)grab((size_t)D_ * C_ * 4);
    float*  bkf   = (float*)grab((size_t)D_ * C_ * 4);
    float*  bvf   = (float*)grab((size_t)D_ * C_ * 4);
    const size_t FH = (size_t)B_ * NKMAX * C_;
    __bf16* f1h  = (__bf16*)grab(FH * 2);
    __bf16* f2h  = (__bf16*)grab(FH * 2);
    __bf16* knb  = (__bf16*)grab(FH * 2);
    __bf16* vnb  = (__bf16*)grab(FH * 2);
    const size_t QE = (size_t)B_ * NQ_ * C_;
    float*  qf    = (float*)grab(QE * 4);
    __bf16* qbf   = (__bf16*)grab(QE * 2);
    __bf16* qhatb = (__bf16*)grab(QE * 2);
    __bf16* qnb   = (__bf16*)grab(QE * 2);
    __bf16* saob  = (__bf16*)grab(QE * 2);
    __bf16* xob   = (__bf16*)grab(QE * 2);
    __bf16* qkvb  = (__bf16*)grab(QE * 3 * 2);
    float*  q1f   = (float*)grab(QE * 4);
    float*  q2f   = (float*)grab(QE * 4);
    float*  r1f   = (float*)grab(QE * 4);
    float*  r2f   = (float*)grab(QE * 4);
    const size_t MSN = (size_t)B_ * NQ_ * NQ_;
    float*  msf  = (float*)grab(MSN * 4);
    float*  sm2f = (float*)grab(MSN * 4);
    float*  sm1f = (float*)grab(MSN * 4);

    auto blks = [](long long n) { return (unsigned)((n + 255) / 256); };

    conv_scale_bf16<<<blks(WCC), 256, 0, stream>>>(ca_wq, ca_ng, wqf, (long long)WCC, C_);
    conv_scale_bf16<<<blks(WCC), 256, 0, stream>>>(ca_wk, ca_ng, wkf, (long long)WCC, C_);
    conv_scale_bf16<<<blks(WCC), 256, 0, stream>>>(ca_wv, ca_ng, wvf, (long long)WCC, C_);
    conv_scale_bf16<<<blks(WCC), 256, 0, stream>>>(ca_wp, nullptr, wpf, (long long)WCC, C_);
    conv_scale_bf16<<<blks(WCC), 256, 0, stream>>>(sa_wp, nullptr, sawpf, (long long)WCC, C_);
    conv_scale_bf16<<<blks(WCC * 3), 256, 0, stream>>>(sa_wqkv, nullptr, sqkvf, (long long)(WCC * 3), 3 * C_);
    fold_bias<<<(D_ * C_ + 255) / 256, 256, 0, stream>>>(ca_nb, ca_wq, bqf);
    fold_bias<<<(D_ * C_ + 255) / 256, 256, 0, stream>>>(ca_nb, ca_wk, bkf);
    fold_bias<<<(D_ * C_ + 255) / 256, 256, 0, stream>>>(ca_nb, ca_wv, bvf);

    auto gemm = [&](const __bf16* A, const __bf16* W, const float* bias,
                    float* Yf, __bf16* Yb, int M, int N) {
        dim3 g((unsigned)((M + 63) / 64), (unsigned)(N / 128));
        gemm_bf16<<<g, 256, 0, stream>>>(A, W, bias, Yf, Yb, M, N, C_);
    };

    const int hw_list[4] = {64, 32, 16, 8};
    const int MROWS = B_ * NQ_;
    const float scale = 0.17677669529663687f;  // 1/sqrt(32)

    for (int l = 0; l < 4; ++l) {
        int hw = hw_list[l];
        int nk2 = hw * hw;
        int MK = B_ * nk2;

        feat_norm<<<(unsigned)(B_ * nk2), 256, 0, stream>>>(feat1[l], f1h, nk2);
        feat_norm<<<(unsigned)(B_ * nk2), 256, 0, stream>>>(feat2[l], f2h, nk2);
        init_q<<<blks(QE), 256, 0, stream>>>(query_embed, qf, (int)QE);

        for (int d = 0; d < D_; ++d) {
            const __bf16* wq_d   = wqf + (size_t)d * C_ * C_;
            const __bf16* wk_d   = wkf + (size_t)d * C_ * C_;
            const __bf16* wv_d   = wvf + (size_t)d * C_ * C_;
            const __bf16* wp_d   = wpf + (size_t)d * C_ * C_;
            const __bf16* sawp_d = sawpf + (size_t)d * C_ * C_;
            const __bf16* sqkv_d = sqkvf + (size_t)d * C_ * 3 * C_;
            const float*  bq_d   = bqf + (size_t)d * C_;
            const float*  bk_d   = bkf + (size_t)d * C_;
            const float*  bv_d   = bvf + (size_t)d * C_;
            const float*  cabp_d = ca_bp + (size_t)d * C_;
            const float*  sabp_d = sa_bp + (size_t)d * C_;

            // ---- self attention (replaces q) ----
            cast_bf16<<<blks(QE), 256, 0, stream>>>(qf, qbf, (int)QE);
            gemm(qbf, sqkv_d, nullptr, nullptr, qkvb, MROWS, 3 * C_);
            mha_flash<<<(unsigned)(B_ * H_), 256, 0, stream>>>(
                qkvb + 0,      3 * C_, (long long)NQ_ * 3 * C_,
                qkvb + C_,     3 * C_, (long long)NQ_ * 3 * C_,
                qkvb + 2 * C_, 3 * C_, (long long)NQ_ * 3 * C_,
                saob, C_, (long long)NQ_ * C_, NQ_, NQ_, scale);
            gemm(saob, sawp_d, sabp_d, qf, nullptr, MROWS, C_);

            // ---- cross attention ----
            ln_rows<<<(unsigned)MROWS, 256, 0, stream>>>(qf, qhatb);
            gemm(qhatb, wq_d, bq_d, nullptr, qnb, MROWS, C_);

            for (int v = 0; v < 2; ++v) {
                const __bf16* fh = v ? f2h : f1h;
                float* qv = v ? q2f : q1f;
                gemm(fh, wk_d, bk_d, nullptr, knb, MK, C_);
                gemm(fh, wv_d, bv_d, nullptr, vnb, MK, C_);
                mha_flash<<<(unsigned)(B_ * H_), 256, 0, stream>>>(
                    qnb, C_, (long long)NQ_ * C_,
                    knb, C_, (long long)nk2 * C_,
                    vnb, C_, (long long)nk2 * C_,
                    xob, C_, (long long)NQ_ * C_, NQ_, nk2, scale);
                gemm(xob, wp_d, cabp_d, qv, nullptr, MROWS, C_);
            }

            // ---- matching + combine ----
            match_scores<<<(unsigned)MROWS, 128, 0, stream>>>(q1f, q2f, msf);
            softmax_ms<<<(unsigned)MROWS, 128, 0, stream>>>(msf, sm2f, 2);
            softmax_ms<<<(unsigned)MROWS, 128, 0, stream>>>(msf, sm1f, 1);
            attn_mix<<<(unsigned)MROWS, 256, 0, stream>>>(q1f, q2f, sm2f, r1f, 0);
            attn_mix<<<(unsigned)MROWS, 256, 0, stream>>>(q2f, q1f, sm1f, r2f, 1);
            combine_softmax<<<(unsigned)MROWS, 256, 0, stream>>>(qf, r1f, r2f);
        }

        hipMemcpyAsync((float*)d_out + (size_t)l * QE, qf, QE * sizeof(float),
                       hipMemcpyDeviceToDevice, stream);
    }
}
